// SparseMoELayer_16544214024521
// MI455X (gfx1250) — matrix operand order repaired and checked by reading, not compiled, not run on hardware
//
#include <hip/hip_runtime.h>
#include <hip/hip_bf16.h>

// ---- problem constants (B,N,D,E,K) = (4,4096,1024,8,2) ----
#define T_TOK 16384            // B*N tokens
#define D_DIM 1024
#define H_DIM 2048             // 2*D
#define E_EXP 8

#define TILE_M 64              // tokens per block tile
#define CHUNK  512             // hidden-dim chunk per pipeline stage
#define NCHUNK (H_DIM / CHUNK) // 4
#define XS_STRIDE 1032         // 1024 + 8 bf16 pad (shifts 4 banks/row)
#define HS_STRIDE 520          // 512 + 8 bf16 pad

typedef __attribute__((ext_vector_type(16))) __bf16 v16bf;
typedef __attribute__((ext_vector_type(8)))  float  v8f;

union Frag {
  v16bf v;
  uint4 q[2];
};

__device__ __forceinline__ unsigned short f2bf(float f) {
  unsigned int u = __float_as_uint(f);
  u += 0x7FFFu + ((u >> 16) & 1u);   // round-to-nearest-even
  return (unsigned short)(u >> 16);
}

// ---------------- weight convert + transpose: f32 [K][N] -> bf16 [N][K] ----
__global__ void convert_w1(const float* __restrict__ W1, unsigned short* __restrict__ W1t) {
  long i = (long)blockIdx.x * blockDim.x + threadIdx.x;     // over E*D*H
  if (i >= (long)E_EXP * D_DIM * H_DIM) return;
  int e = (int)(i / ((long)D_DIM * H_DIM));
  int rem = (int)(i % ((long)D_DIM * H_DIM));
  int k = rem / H_DIM;            // input dim
  int n = rem % H_DIM;            // output col
  W1t[(long)e * H_DIM * D_DIM + (long)n * D_DIM + k] = f2bf(W1[i]);
}

__global__ void convert_w2(const float* __restrict__ W2, unsigned short* __restrict__ W2t) {
  long i = (long)blockIdx.x * blockDim.x + threadIdx.x;     // over E*H*D
  if (i >= (long)E_EXP * H_DIM * D_DIM) return;
  int e = (int)(i / ((long)H_DIM * D_DIM));
  int rem = (int)(i % ((long)H_DIM * D_DIM));
  int k = rem / D_DIM;            // input dim (2048)
  int n = rem % D_DIM;            // output col (1024)
  W2t[(long)e * D_DIM * H_DIM + (long)n * H_DIM + k] = f2bf(W2[i]);
}

// ---------------- gating: logits = x@Wg+bg, top-2 softmax, build expert lists
__global__ void gate_kernel(const float* __restrict__ x, const float* __restrict__ Wg,
                            const float* __restrict__ bg, int* __restrict__ cnt,
                            int* __restrict__ tok, float* __restrict__ gw) {
  int wave = threadIdx.x >> 5;
  int lane = threadIdx.x & 31;
  int t = blockIdx.x * (blockDim.x >> 5) + wave;
  if (t >= T_TOK) return;
  const float* xr = x + (long)t * D_DIM;
  float acc[E_EXP];
#pragma unroll
  for (int e = 0; e < E_EXP; ++e) acc[e] = 0.f;
  for (int d = lane; d < D_DIM; d += 32) {
    float xv = xr[d];
    const float* wr = Wg + d * E_EXP;
#pragma unroll
    for (int e = 0; e < E_EXP; ++e) acc[e] += xv * wr[e];
  }
#pragma unroll
  for (int off = 16; off >= 1; off >>= 1) {
#pragma unroll
    for (int e = 0; e < E_EXP; ++e) acc[e] += __shfl_xor(acc[e], off, 32);
  }
  if (lane == 0) {
    float l[E_EXP];
#pragma unroll
    for (int e = 0; e < E_EXP; ++e) l[e] = acc[e] + bg[e];
    int i1 = 0;
#pragma unroll
    for (int e = 1; e < E_EXP; ++e) if (l[e] > l[i1]) i1 = e;
    int i2 = (i1 == 0) ? 1 : 0;
#pragma unroll
    for (int e = 0; e < E_EXP; ++e) if (e != i1 && l[e] > l[i2]) i2 = e;
    float m  = l[i1];
    float e1 = __expf(l[i1] - m), e2 = __expf(l[i2] - m);
    float s  = e1 + e2;
    int s1 = atomicAdd(&cnt[i1], 1);
    tok[i1 * T_TOK + s1] = t; gw[i1 * T_TOK + s1] = e1 / s;
    int s2 = atomicAdd(&cnt[i2], 1);
    tok[i2 * T_TOK + s2] = t; gw[i2 * T_TOK + s2] = e2 / s;
  }
}

// ---------------- fused per-expert FFN: y = relu(xg@W1+b1)@W2+b2, out += g*y
__global__ void __launch_bounds__(512)
moe_expert_kernel(const float* __restrict__ x,
                  const unsigned short* __restrict__ W1t,  // [E][H][D] bf16
                  const unsigned short* __restrict__ W2t,  // [E][D][H] bf16
                  const float* __restrict__ b1,
                  const float* __restrict__ b2,
                  const int* __restrict__ cnt,
                  const int* __restrict__ tok,
                  const float* __restrict__ gw,
                  float* __restrict__ out) {
  const int e = blockIdx.y;
  const int tileStart = blockIdx.x * TILE_M;
  const int c = cnt[e];
  if (tileStart >= c) return;                 // uniform branch, EXEC stays full

  extern __shared__ char smemraw[];
  unsigned short* xs = (unsigned short*)smemraw;               // [64][XS_STRIDE]
  unsigned short* hs = xs + TILE_M * XS_STRIDE;                // [64][HS_STRIDE]
  int*   stok = (int*)(hs + TILE_M * HS_STRIDE);
  float* sg   = (float*)(stok + TILE_M);

  const int tid  = threadIdx.x;
  const int wv   = tid >> 5;          // wave 0..15
  const int lane = tid & 31;
  const int hf   = lane >> 4;         // half-wave (WMMA lane group)
  const int lm   = lane & 15;

  if (tid < TILE_M) {
    int r = tileStart + tid;
    if (r < c) { stok[tid] = tok[e * T_TOK + r]; sg[tid] = gw[e * T_TOK + r]; }
    else       { stok[tid] = 0;                  sg[tid] = 0.f; }
  }
  __syncthreads();

  // x tile -> bf16 LDS, zero-pad invalid rows
  for (int i = tid; i < TILE_M * (D_DIM / 4); i += 512) {
    int row = i >> 8;                 // D_DIM/4 = 256 float4 per row
    int c4  = i & 255;
    float4 v = make_float4(0.f, 0.f, 0.f, 0.f);
    if (tileStart + row < c) v = ((const float4*)(x + (long)stok[row] * D_DIM))[c4];
    unsigned int lo = ((unsigned int)f2bf(v.y) << 16) | f2bf(v.x);
    unsigned int hi = ((unsigned int)f2bf(v.w) << 16) | f2bf(v.z);
    *(uint2*)(xs + row * XS_STRIDE + c4 * 4) = make_uint2(lo, hi);
  }

  const unsigned short* w1e = W1t + (long)e * H_DIM * D_DIM;
  const unsigned short* w2e = W2t + (long)e * D_DIM * H_DIM;
  const float* b1e = b1 + e * H_DIM;
  const float* b2e = b2 + e * D_DIM;

  v8f zero = {0.f, 0.f, 0.f, 0.f, 0.f, 0.f, 0.f, 0.f};
  v8f yacc[16];                        // wave owns out cols [wv*64, wv*64+64)
#pragma unroll
  for (int i = 0; i < 16; ++i) yacc[i] = zero;

  __syncthreads();

  for (int ch = 0; ch < NCHUNK; ++ch) {
    // ---- GEMM1: h[64][512] = relu(xs @ W1[:, ch*512..) + b1)
    //      wave owns local cols [wv*32, wv*32+32)  -> 2 N-tiles x 4 M-tiles
    v8f c1[8];
#pragma unroll
    for (int i = 0; i < 8; ++i) c1[i] = zero;
    const int ngBase = ch * CHUNK + wv * 32 + lm;              // +nt*16 per tile
    for (int ks = 0; ks < 32; ++ks) {                          // K = 1024 / 32
      Frag a[4];                                               // hoisted: distinct regs
#pragma unroll
      for (int mt = 0; mt < 4; ++mt) {
        const unsigned short* ap = xs + (mt * 16 + lm) * XS_STRIDE + ks * 32 + hf * 16;
        a[mt].q[0] = *(const uint4*)(ap);
        a[mt].q[1] = *(const uint4*)(ap + 8);
      }
#pragma unroll
      for (int nt = 0; nt < 2; ++nt) {
        Frag b;
        const uint4* bq =
            (const uint4*)(w1e + (long)(ngBase + nt * 16) * D_DIM + ks * 32 + hf * 16);
        b.q[0] = bq[0];
        b.q[1] = bq[1];
#pragma unroll
        for (int mt = 0; mt < 4; ++mt) {
          c1[nt * 4 + mt] = __builtin_amdgcn_wmma_f32_16x16x32_bf16(
              false, a[mt].v, false, b.v, (short)0, c1[nt * 4 + mt], false, false);
        }
      }
    }
    __syncthreads();                   // previous chunk's GEMM2 done reading hs
#pragma unroll
    for (int nt = 0; nt < 2; ++nt) {
      float bias = b1e[ngBase + nt * 16];
#pragma unroll
      for (int mt = 0; mt < 4; ++mt) {
#pragma unroll
        for (int r = 0; r < 8; ++r) {
          float hv = c1[nt * 4 + mt][r] + bias;
          hv = hv > 0.f ? hv : 0.f;
          hs[(mt * 16 + hf * 8 + r) * HS_STRIDE + wv * 32 + nt * 16 + lm] = f2bf(hv);
        }
      }
    }
    __syncthreads();                   // hs ready

    // ---- GEMM2: y[64][1024] += h[64][512] @ W2[ch*512.., :]
    for (int ks = 0; ks < 16; ++ks) {  // K = 512 / 32
      Frag a[4];
#pragma unroll
      for (int mt = 0; mt < 4; ++mt) {
        const unsigned short* ap = hs + (mt * 16 + lm) * HS_STRIDE + ks * 32 + hf * 16;
        a[mt].q[0] = *(const uint4*)(ap);
        a[mt].q[1] = *(const uint4*)(ap + 8);
      }
#pragma unroll
      for (int nt = 0; nt < 4; ++nt) {
        const int n = wv * 64 + nt * 16 + lm;
        const uint4* bq =
            (const uint4*)(w2e + (long)n * H_DIM + ch * CHUNK + ks * 32 + hf * 16);
        Frag b;
        b.q[0] = bq[0];
        b.q[1] = bq[1];
#pragma unroll
        for (int mt = 0; mt < 4; ++mt) {
          yacc[mt * 4 + nt] = __builtin_amdgcn_wmma_f32_16x16x32_bf16(
              false, a[mt].v, false, b.v, (short)0, yacc[mt * 4 + nt], false, false);
        }
      }
    }
  }

  // ---- epilogue: +b2, gate scale, accumulate into out (token gets 2 experts)
#pragma unroll
  for (int nt = 0; nt < 4; ++nt) {
    const int col = wv * 64 + nt * 16 + lm;
    float bias2 = b2e[col];
#pragma unroll
    for (int mt = 0; mt < 4; ++mt) {
#pragma unroll
      for (int r = 0; r < 8; ++r) {
        int row = mt * 16 + hf * 8 + r;
        if (tileStart + row < c) {
          float val = sg[row] * (yacc[mt * 4 + nt][r] + bias2);
          atomicAdd(out + (long)stok[row] * D_DIM + col, val);
        }
      }
    }
  }
}

// ---------------- host-side launch ----------------
extern "C" void kernel_launch(void* const* d_in, const int* in_sizes, int n_in,
                              void* d_out, int out_size, void* d_ws, size_t ws_size,
                              hipStream_t stream) {
  (void)in_sizes; (void)n_in; (void)ws_size;
  const float* x  = (const float*)d_in[0];
  const float* Wg = (const float*)d_in[1];
  const float* bg = (const float*)d_in[2];
  const float* W1 = (const float*)d_in[3];
  const float* b1 = (const float*)d_in[4];
  const float* W2 = (const float*)d_in[5];
  const float* b2 = (const float*)d_in[6];
  float* out = (float*)d_out;

  // workspace layout (bytes): W1t 32MB | W2t 32MB | cnt 256B | tok 512KB | gw 512KB
  char* ws = (char*)d_ws;
  unsigned short* W1t = (unsigned short*)ws;
  unsigned short* W2t = (unsigned short*)(ws + 33554432);
  int*   cnt = (int*)  (ws + 67108864);
  int*   tokl= (int*)  (ws + 67109120);
  float* gwl = (float*)(ws + 67633408);

  hipMemsetAsync(cnt, 0, 256, stream);
  hipMemsetAsync(d_out, 0, (size_t)out_size * sizeof(float), stream);

  {
    long n = (long)E_EXP * D_DIM * H_DIM;
    convert_w1<<<dim3((unsigned)((n + 255) / 256)), dim3(256), 0, stream>>>(W1, W1t);
  }
  {
    long n = (long)E_EXP * H_DIM * D_DIM;
    convert_w2<<<dim3((unsigned)((n + 255) / 256)), dim3(256), 0, stream>>>(W2, W2t);
  }
  gate_kernel<<<dim3(T_TOK / 8), dim3(256), 0, stream>>>(x, Wg, bg, cnt, tokl, gwl);

  size_t shmem = (size_t)TILE_M * XS_STRIDE * 2 + (size_t)TILE_M * HS_STRIDE * 2
               + (size_t)TILE_M * 8;   // xs + hs + stok + sg  (~195 KB of 320 KB LDS)
  dim3 grid(T_TOK / TILE_M, E_EXP);    // worst case: every token routed to one expert
  moe_expert_kernel<<<grid, dim3(512), shmem, stream>>>(
      x, W1t, W2t, b1, b2, cnt, tokl, gwl, out);
}
